// E63mMatrixNonlinearCell_4887672783111
// MI455X (gfx1250) — compile-verified, matrix operand order checked by reading, not run on hardware
//
#include <hip/hip_runtime.h>
#include <math.h>

// ---------------------------------------------------------------------------
// Problem constants (T=256, B=8, D=256, N=256)
// ---------------------------------------------------------------------------
#define T_STEPS 256
#define BATCH   8
#define DDIM    256
#define NDIM    256
#define CHUNKS  8               // workgroups per batch (split along N rows)
#define ROWS    (NDIM / CHUNKS) // 32 rows of state per workgroup
#define SCAN_THREADS 256
#define TB_ROWS (T_STEPS * BATCH) // 2048 rows for the projection GEMMs

typedef _Float16 v16h __attribute__((ext_vector_type(16)));
typedef float    v8f  __attribute__((ext_vector_type(8)));
typedef float    f4   __attribute__((ext_vector_type(4)));

// ---------------------------------------------------------------------------
// Kernel 0: reset inter-workgroup barrier state (must run every replay)
// ---------------------------------------------------------------------------
__global__ void init_kernel(int* __restrict__ cnt, int* __restrict__ phase) {
    int i = threadIdx.x;
    if (i < BATCH * 32) { cnt[i] = 0; phase[i] = 0; }
}

// ---------------------------------------------------------------------------
// Kernel 1: fused projection GEMMs via WMMA f32<-f16 16x16x32.
//   out[w] = X[2048,256] @ W_w^T   for w in {W_k, W_q, W_x, W_alpha}
// One 16x16 output tile per wave32; fully unrolled K-loop of 8
// v_wmma_f32_16x16x32_f16. Fragment layouts follow CDNA5 ISA 7.12.2
// (16-bit A 16x32 / B 32x16): each lane holds two contiguous 8-half runs
// at k0+ko and k0+16+ko, ko=(lane/16)*8.
// ---------------------------------------------------------------------------
__global__ __launch_bounds__(256) void proj_kernel(
    const float* __restrict__ x,
    const float* __restrict__ Wk, const float* __restrict__ Wq,
    const float* __restrict__ Wx, const float* __restrict__ Wa,
    float* __restrict__ out)      // 4 consecutive [2048,256] arrays
{
    const int wave = (int)((blockIdx.x * blockDim.x + threadIdx.x) >> 5);
    const int lane = threadIdx.x & 31;
    const int w  = wave & 3;          // which weight matrix
    const int nt = (wave >> 2) & 15;  // n-tile (16 tiles of 16)
    const int mt = wave >> 6;         // m-tile (128 tiles of 16)

    const float* W = (w == 0) ? Wk : (w == 1) ? Wq : (w == 2) ? Wx : Wa;
    float* dst = out + (size_t)w * ((size_t)TB_ROWS * NDIM);

    const int hi = lane >> 4;   // half-wave select
    const int ml = lane & 15;   // row-in-tile (A) / col-in-tile (B,C)
    const int ko = hi * 8;

    const float* xrow = x + (size_t)(mt * 16 + ml) * DDIM;
    const float* wrow = W + (size_t)(nt * 16 + ml) * DDIM; // B[k][n] = W[n][k]

    v8f c = {};
    #pragma unroll
    for (int k0 = 0; k0 < DDIM; k0 += 32) {
        v16h a, b;
        const int ka = k0 + ko;
        #pragma unroll
        for (int e = 0; e < 8; ++e) {
            a[e]     = (_Float16)xrow[ka + e];
            a[e + 8] = (_Float16)xrow[ka + 16 + e];
            b[e]     = (_Float16)wrow[ka + e];
            b[e + 8] = (_Float16)wrow[ka + 16 + e];
        }
        c = __builtin_amdgcn_wmma_f32_16x16x32_f16(
                /*neg_a=*/false, a, /*neg_b=*/false, b,
                /*c_mod=*/(short)0, c, /*reuse_a=*/false, /*reuse_b=*/false);
    }

    // C/D layout: lane ml = N, VGPR r -> M = r + 8*hi
    const int n = nt * 16 + ml;
    #pragma unroll
    for (int r = 0; r < 8; ++r) {
        const int m = mt * 16 + r + 8 * hi;
        dst[(size_t)m * NDIM + n] = c[r];
    }
}

// ---------------------------------------------------------------------------
// Kernel 2: sequential scan. Grid = B*CHUNKS = 64 workgroups.
// WG (b,c) owns rows [c*32, c*32+32) of S[b] in LDS (32 KB).
// Per step: async-load k/q rows straight into LDS (ASYNCcnt path) ->
// local GEMV partials -> all-gather `retrieved` across the 8 WGs of batch b
// (double-buffered global staging + atomic arrive/phase barrier) ->
// v/alpha -> rank-1 state update (nontemporal b128 stores of S history) -> y.
// ---------------------------------------------------------------------------
__global__ __launch_bounds__(SCAN_THREADS) void scan_kernel(
    const float* __restrict__ S0,      // [B,N,D]
    const float* __restrict__ Wr,      // [N,N]
    const float* __restrict__ bvec,    // [N]
    const float* __restrict__ balpha,  // [N]
    const float* __restrict__ kall,    // [T*B,D]
    const float* __restrict__ qall,    // [T*B,D]
    const float* __restrict__ wxall,   // [T*B,N]
    const float* __restrict__ axall,   // [T*B,N]
    float* __restrict__ y_out,         // [T,B,N]
    float* __restrict__ S_out,         // [T+1,B,N,D]
    int*   __restrict__ cnt,           // [B*32]
    int*   __restrict__ phase,         // [B*32]
    float* __restrict__ rbuf)          // [2,B,N]
{
    __shared__ __align__(16) float S_l[ROWS][DDIM]; // 32 KB state slice
    __shared__ __align__(16) float k_l[DDIM];
    __shared__ __align__(16) float q_l[DDIM];
    __shared__ float r_l[NDIM];
    __shared__ float red[SCAN_THREADS];
    __shared__ float v_l[ROWS];
    __shared__ float a_l[ROWS];

    const int tid = threadIdx.x;
    const int b   = blockIdx.x >> 3;
    const int c   = blockIdx.x & 7;
    const int r   = tid >> 3;   // 0..31 local row
    const int p   = tid & 7;    // 0..7  column slice
    const int cb  = p * 32;     // column base

    int* mycnt   = cnt   + b * 32;
    int* myphase = phase + b * 32;

    // LDS byte offsets of the k/q staging buffers (flat LDS aperture:
    // addr[31:0] is the wave-relative LDS offset).
    const unsigned kdst = (unsigned)(uintptr_t)(&k_l[tid]);
    const unsigned qdst = (unsigned)(uintptr_t)(&q_l[tid]);

    // --- load S0 slice into LDS and emit S_hist[0] (== S0) ---
    {
        const float* s0 = S0 + ((size_t)b * NDIM + (size_t)c * ROWS) * DDIM;
        float* h0 = S_out + (size_t)b * NDIM * DDIM + (size_t)c * ROWS * DDIM;
        for (int i = tid; i < ROWS * DDIM / 4; i += SCAN_THREADS) {
            f4 v = ((const f4*)s0)[i];
            ((f4*)&S_l[0][0])[i] = v;
            __builtin_nontemporal_store(v, ((f4*)h0) + i);
        }
    }
    __syncthreads();

    for (int t = 0; t < T_STEPS; ++t) {
        const size_t row = (size_t)t * BATCH + b;

        // stage k_t, q_t via async global->LDS (ASYNCcnt, no VGPR round trip)
        {
            const float* ksrc = &kall[row * DDIM + tid];
            const float* qsrc = &qall[row * DDIM + tid];
            asm volatile("global_load_async_to_lds_b32 %0, %1, off"
                         :: "v"(kdst), "v"(ksrc) : "memory");
            asm volatile("global_load_async_to_lds_b32 %0, %1, off"
                         :: "v"(qdst), "v"(qsrc) : "memory");
        }
        if (t + 1 < T_STEPS)
            __builtin_prefetch(&kall[(row + BATCH) * DDIM + tid], 0, 1);
        asm volatile("s_wait_asynccnt 0x0" ::: "memory");
        __syncthreads();

        // --- Sk = S_slice @ k_t  (8 partials per row, LDS reduce) ---
        float acc = 0.f;
        #pragma unroll 8
        for (int i = 0; i < 32; ++i) acc += S_l[r][cb + i] * k_l[cb + i];
        red[tid] = acc;
        __syncthreads();
        if (tid < ROWS) {
            float s = 0.f;
            #pragma unroll
            for (int i = 0; i < 8; ++i) s += red[tid * 8 + i];
            rbuf[(size_t)(t & 1) * BATCH * NDIM + (size_t)b * NDIM + c * ROWS + tid]
                = tanhf(s);
        }
        __threadfence();   // make retrieved chunk agent-visible before arrive
        __syncthreads();

        // --- 8-WG barrier for batch b (monotonic count + phase) ---
        if (tid == 0) {
            int old = __hip_atomic_fetch_add(mycnt, 1, __ATOMIC_ACQ_REL,
                                             __HIP_MEMORY_SCOPE_AGENT);
            if (old == 8 * t + 7) {
                __hip_atomic_fetch_add(myphase, 1, __ATOMIC_ACQ_REL,
                                       __HIP_MEMORY_SCOPE_AGENT);
            } else {
                while (__hip_atomic_load(myphase, __ATOMIC_ACQUIRE,
                                         __HIP_MEMORY_SCOPE_AGENT) <= t) {
                    __builtin_amdgcn_s_sleep(2);
                }
            }
        }
        __syncthreads();

        // --- gather full retrieved[N] ---
        r_l[tid] = rbuf[(size_t)(t & 1) * BATCH * NDIM + (size_t)b * NDIM + tid];
        __syncthreads();

        // --- v = tanh(W_r @ retrieved + Wx + b); alpha = sigmoid(ax + b_a) ---
        {
            const float* wr = Wr + (size_t)(c * ROWS + r) * NDIM + cb;
            float vacc = 0.f;
            #pragma unroll 8
            for (int i = 0; i < 32; ++i) vacc += wr[i] * r_l[cb + i];
            red[tid] = vacc;
        }
        __syncthreads();
        if (tid < ROWS) {
            const int n = c * ROWS + tid;
            float s = 0.f;
            #pragma unroll
            for (int i = 0; i < 8; ++i) s += red[tid * 8 + i];
            v_l[tid] = tanhf(s + wxall[row * NDIM + n] + bvec[n]);
            a_l[tid] = 1.f / (1.f + __expf(-(axall[row * NDIM + n] + balpha[n])));
        }
        __syncthreads();

        // --- S = a*S + (1-a)*v⊗k ; stream S_hist[t+1]; y partials ---
        {
            const float al = a_l[r];
            const float om = (1.f - al) * v_l[r];
            float* hrow = S_out
                + ((size_t)(t + 1) * BATCH + b) * (size_t)NDIM * DDIM
                + (size_t)(c * ROWS + r) * DDIM + cb;
            float qacc = 0.f;
            #pragma unroll
            for (int i = 0; i < 8; ++i) {
                f4 s  = ((f4*)&S_l[r][cb])[i];
                f4 kk = ((f4*)&k_l[cb])[i];
                s = al * s + om * kk;
                ((f4*)&S_l[r][cb])[i] = s;
                __builtin_nontemporal_store(s, ((f4*)hrow) + i);
                f4 qq = ((f4*)&q_l[cb])[i];
                qacc += s[0]*qq[0] + s[1]*qq[1] + s[2]*qq[2] + s[3]*qq[3];
            }
            red[tid] = qacc;
        }
        __syncthreads();
        if (tid < ROWS) {
            float s = 0.f;
            #pragma unroll
            for (int i = 0; i < 8; ++i) s += red[tid * 8 + i];
            y_out[row * NDIM + c * ROWS + tid] = tanhf(s);
        }
        __syncthreads(); // red/k_l/q_l are rewritten next iteration
    }
}

// ---------------------------------------------------------------------------
// Host launcher
// ---------------------------------------------------------------------------
extern "C" void kernel_launch(void* const* d_in, const int* in_sizes, int n_in,
                              void* d_out, int out_size, void* d_ws, size_t ws_size,
                              hipStream_t stream) {
    const float* x  = (const float*)d_in[0];
    const float* S0 = (const float*)d_in[1];
    const float* Wk = (const float*)d_in[2];
    const float* Wq = (const float*)d_in[3];
    const float* Wx = (const float*)d_in[4];
    const float* Wr = (const float*)d_in[5];
    const float* bv = (const float*)d_in[6];
    const float* Wa = (const float*)d_in[7];
    const float* ba = (const float*)d_in[8];

    char* ws = (char*)d_ws;
    int*   cnt   = (int*)ws;                 // B*32 ints
    int*   phase = (int*)(ws + 1024);        // B*32 ints
    float* rbuf  = (float*)(ws + 2048);      // 2*B*N floats (16 KB)
    float* proj  = (float*)(ws + 32768);     // 4 * 2048*256 floats (8 MB)

    const size_t PROJ = (size_t)TB_ROWS * NDIM;
    float* kall  = proj;
    float* qall  = proj + PROJ;
    float* wxall = proj + 2 * PROJ;
    float* axall = proj + 3 * PROJ;

    float* y_out = (float*)d_out;                            // [T,B,N]
    float* S_out = y_out + (size_t)T_STEPS * BATCH * NDIM;   // [T+1,B,N,D]

    hipLaunchKernelGGL(init_kernel, dim3(1), dim3(256), 0, stream, cnt, phase);

    // 8192 output tiles (128 m-tiles * 16 n-tiles * 4 matrices), 1 per wave,
    // 8 waves per 256-thread block -> 1024 blocks.
    hipLaunchKernelGGL(proj_kernel, dim3(1024), dim3(256), 0, stream,
                       x, Wk, Wq, Wx, Wa, proj);

    hipLaunchKernelGGL(scan_kernel, dim3(BATCH * CHUNKS), dim3(SCAN_THREADS), 0, stream,
                       S0, Wr, bv, ba, kall, qall, wxall, axall,
                       y_out, S_out, cnt, phase, rbuf);
}
